// Encoder_36524401885236
// MI455X (gfx1250) — compile-verified, matrix operand order checked by reading, not run on hardware
//
#include <hip/hip_runtime.h>
#include <hip/hip_bf16.h>

#define HH 32
#define ZZ 32
#define GG 64
#define NLAYERS 4
#define WPB 8   // waves per 256-thread block (wave32)

typedef __attribute__((ext_vector_type(16))) __bf16 bf16x16;
typedef __attribute__((ext_vector_type(8)))  float  f32x8;

union Frag {
  bf16x16 bf;
  unsigned short u[16];
};

__device__ __forceinline__ unsigned short f2bf(float f) {
  union { float f; unsigned u; } v; v.f = f;
  unsigned r = v.u + 0x7FFFu + ((v.u >> 16) & 1u);   // RNE
  return (unsigned short)(r >> 16);
}
__device__ __forceinline__ float bf2f(unsigned short s) {
  union { unsigned u; float f; } v; v.u = ((unsigned)s) << 16;
  return v.f;
}
// native v_rcp_f32 (TRANS op, co-executes with WMMA) instead of IEEE divide
__device__ __forceinline__ float frcp(float x) { return __builtin_amdgcn_rcpf(x); }
__device__ __forceinline__ float silu_f(float x) {
  return x * frcp(1.0f + __expf(-x));
}
__device__ __forceinline__ float tanh_fast(float x) {
  // 1 - 2/(e^{2x}+1); saturates to +/-1 correctly for large |x|
  float e = __expf(2.0f * x);
  return 1.0f - 2.0f * frcp(e + 1.0f);
}
// 16-bit A-matrix 16x32 per-lane K index (ISA 7.12.2)
__device__ __forceinline__ int ak_of(int lane, int i) {
  int r = i >> 1;
  return ((r >> 2) << 4) + ((r & 3) << 1) + (i & 1) + ((lane & 16) ? 8 : 0);
}
__device__ __forceinline__ f32x8 wmma_bf16(Frag a, Frag b, f32x8 c) {
  return __builtin_amdgcn_wmma_f32_16x16x32_bf16(false, a.bf, false, b.bf,
                                                 (short)0, c, false, false);
}
// Load a 32(K) x 16(N) B-fragment from a row-major [32 x 32] bf16 weight chunk.
__device__ __forceinline__ void load_bfrag(Frag* f, const unsigned short* __restrict__ w,
                                           int lane, int nh) {
  int n = (lane & 15) + nh * 16;
#pragma unroll
  for (int i = 0; i < 16; ++i) {
    int k = ak_of(lane, i);
    f->u[i] = w[k * HH + n];
  }
}

// ---------------------------------------------------------------- utility kernels
__global__ void cvt_bf16_kernel(const float* __restrict__ src,
                                unsigned short* __restrict__ dst, int n) {
  int i = blockIdx.x * blockDim.x + threadIdx.x;
  if (i < n) dst[i] = f2bf(src[i]);
}

__global__ void egnn_embed_kernel(const float* __restrict__ x,
                                  const float* __restrict__ pos,
                                  const float* __restrict__ eiw,
                                  const float* __restrict__ eib,
                                  float* __restrict__ h,
                                  unsigned short* __restrict__ hbf,
                                  float* __restrict__ coord, int N) {
  int idx = blockIdx.x * blockDim.x + threadIdx.x;
  if (idx < N * HH) {
    int j = idx & (HH - 1);
    int v = idx >> 5;
    float val = x[v] * eiw[j] + eib[j];
    h[idx] = val;
    hbf[idx] = f2bf(val);
  }
  if (idx < N * 3) coord[idx] = pos[idx];
}

__global__ void egnn_deg_kernel(const int* __restrict__ row,
                                float* __restrict__ deg, int E) {
  int e = blockIdx.x * blockDim.x + threadIdx.x;
  if (e < E) atomicAdd(&deg[row[e]], 1.0f);
}

// ---------------------------------------------------------------- edge stage
__global__ void __launch_bounds__(256, 1)
egnn_edge_kernel(const unsigned short* __restrict__ hbf,
                 const float* __restrict__ coord,
                 const int* __restrict__ row,
                 const int* __restrict__ col,
                 const float* __restrict__ lcell,
                 const unsigned short* __restrict__ w1bf,  // [65,32] bf16 (rows 0..63 used)
                 const float* __restrict__ w1last,         // row 64 of edge_w1 (f32)
                 const float* __restrict__ b1,
                 const unsigned short* __restrict__ w2bf,  // [32,32]
                 const float* __restrict__ b2,
                 const unsigned short* __restrict__ cw1bf, // [32,32]
                 const float* __restrict__ cb1,
                 const float* __restrict__ cw2,            // [32] f32
                 const float* __restrict__ cb2,            // [1]
                 float* __restrict__ magg,
                 float* __restrict__ tsum,
                 int E, int nTiles) {
  __shared__ unsigned short lds_m[WPB][16 * HH];
  __shared__ float lds_rad[WPB][16];

  const int lane = threadIdx.x & 31;
  const int wid  = threadIdx.x >> 5;
  const int n    = lane & 15;
  const int hi   = (lane & 16) ? 1 : 0;
  const int m    = n;

  // Per-wave weight fragments, loaded once for the whole grid-stride loop.
  Frag b1f[4], b2f[2], c1f[2];
#pragma unroll
  for (int kc = 0; kc < 2; ++kc)
#pragma unroll
    for (int nh = 0; nh < 2; ++nh)
      load_bfrag(&b1f[kc * 2 + nh], w1bf + kc * 32 * HH, lane, nh);
#pragma unroll
  for (int nh = 0; nh < 2; ++nh) {
    load_bfrag(&b2f[nh], w2bf, lane, nh);
    load_bfrag(&c1f[nh], cw1bf, lane, nh);
  }
  const float bias1lo = b1[n],  bias1hi = b1[n + 16];
  const float bias2lo = b2[n],  bias2hi = b2[n + 16];
  const float cb1lo   = cb1[n], cb1hi   = cb1[n + 16];
  const float w1llo = w1last[n], w1lhi = w1last[n + 16];
  const float cb2s = cb2[0];
  const float Lx = lcell[0], Ly = lcell[1], Lz = lcell[2];
  const float iLx = frcp(Lx), iLy = frcp(Ly), iLz = frcp(Lz);

  const int stride = gridDim.x * WPB;
  const int iters  = (nTiles + stride - 1) / stride;
  int t = blockIdx.x * WPB + wid;

  for (int it = 0; it < iters; ++it, t += stride) {
    const bool active = (t < nTiles);
    const int e0 = (active ? t : 0) * 16;

    if (lane == 0) {
      int tn = t + stride;
      if (tn < nTiles) {
        __builtin_prefetch(row + tn * 16, 0, 0);
        __builtin_prefetch(col + tn * 16, 0, 0);
      }
    }

    // geometry: radial + normalized min-image diff (lanes 0..15, one edge each)
    float cdx = 0.f, cdy = 0.f, cdz = 0.f;
    if (lane < 16) {
      int e = e0 + lane; if (e >= E) e = E - 1;
      int rv = row[e], cv = col[e];
      float dx = coord[rv * 3 + 0] - coord[cv * 3 + 0];
      float dy = coord[rv * 3 + 1] - coord[cv * 3 + 1];
      float dz = coord[rv * 3 + 2] - coord[cv * 3 + 2];
      dx -= Lx * rintf(dx * iLx);
      dy -= Ly * rintf(dy * iLy);
      dz -= Lz * rintf(dz * iLz);
      float rad = dx * dx + dy * dy + dz * dz;
      float inv = frcp(sqrtf(rad) + 1e-8f);
      cdx = dx * inv; cdy = dy * inv; cdz = dz * inv;
      lds_rad[wid][lane] = rad;
    }
    __syncthreads();

    // A fragments: gathered bf16 features of h[row] and h[col]
    int e = e0 + m; if (e >= E) e = E - 1;
    const int rv = row[e], cv = col[e];
    Frag ar, ac;
#pragma unroll
    for (int i = 0; i < 16; ++i) {
      int k = ak_of(lane, i);
      ar.u[i] = hbf[rv * HH + k];
      ac.u[i] = hbf[cv * HH + k];
    }

    // GEMM1: [hrow|hcol] x W1[0:64,:]  + bias + rank-1 radial term (row 64)
    f32x8 c0, c1;
#pragma unroll
    for (int r = 0; r < 8; ++r) {
      float rad = lds_rad[wid][r + hi * 8];
      c0[r] = bias1lo + rad * w1llo;
      c1[r] = bias1hi + rad * w1lhi;
    }
    c0 = wmma_bf16(ar, b1f[0], c0);
    c1 = wmma_bf16(ar, b1f[1], c1);
    c0 = wmma_bf16(ac, b1f[2], c0);
    c1 = wmma_bf16(ac, b1f[3], c1);

#pragma unroll
    for (int r = 0; r < 8; ++r) {
      int M = r + hi * 8;
      lds_m[wid][M * HH + n]      = f2bf(silu_f(c0[r]));
      lds_m[wid][M * HH + 16 + n] = f2bf(silu_f(c1[r]));
    }
    __syncthreads();

    // GEMM2: m = silu(m1 @ W2 + b2); scatter magg
    Frag a2;
#pragma unroll
    for (int i = 0; i < 16; ++i) a2.u[i] = lds_m[wid][m * HH + ak_of(lane, i)];
    f32x8 d0, d1;
#pragma unroll
    for (int r = 0; r < 8; ++r) { d0[r] = bias2lo; d1[r] = bias2hi; }
    d0 = wmma_bf16(a2, b2f[0], d0);
    d1 = wmma_bf16(a2, b2f[1], d1);
    __syncthreads();

#pragma unroll
    for (int r = 0; r < 8; ++r) {
      int M = r + hi * 8;
      float v0 = silu_f(d0[r]);
      float v1 = silu_f(d1[r]);
      lds_m[wid][M * HH + n]      = f2bf(v0);
      lds_m[wid][M * HH + 16 + n] = f2bf(v1);
      int ee = e0 + M;
      if (active && ee < E) {
        int rr = row[ee];
        atomicAdd(&magg[rr * HH + n],      v0);
        atomicAdd(&magg[rr * HH + 16 + n], v1);
      }
    }
    __syncthreads();

    // GEMM3: phi1 = silu(m @ Cw1 + cb1)
    Frag a3;
#pragma unroll
    for (int i = 0; i < 16; ++i) a3.u[i] = lds_m[wid][m * HH + ak_of(lane, i)];
    f32x8 p0, p1;
#pragma unroll
    for (int r = 0; r < 8; ++r) { p0[r] = cb1lo; p1[r] = cb1hi; }
    p0 = wmma_bf16(a3, c1f[0], p0);
    p1 = wmma_bf16(a3, c1f[1], p1);

#pragma unroll
    for (int r = 0; r < 8; ++r) {
      int M = r + hi * 8;
      lds_m[wid][M * HH + n]      = f2bf(silu_f(p0[r]));
      lds_m[wid][M * HH + 16 + n] = f2bf(silu_f(p1[r]));
    }
    __syncthreads();

    // phi = tanh(phi1 @ cw2 + cb2); scatter tsum
    if (lane < 16) {
      int ee = e0 + lane;
      if (active && ee < E) {
        float phi = cb2s;
#pragma unroll
        for (int k = 0; k < HH; ++k)
          phi += bf2f(lds_m[wid][lane * HH + k]) * cw2[k];
        phi = tanh_fast(phi);
        int rr = row[ee];
        atomicAdd(&tsum[rr * 3 + 0], cdx * phi);
        atomicAdd(&tsum[rr * 3 + 1], cdy * phi);
        atomicAdd(&tsum[rr * 3 + 2], cdz * phi);
      }
    }
    __syncthreads();
  }
}

// ---------------------------------------------------------------- node stage
__global__ void __launch_bounds__(256, 1)
egnn_node_kernel(float* __restrict__ h,
                 unsigned short* __restrict__ hbf,
                 float* __restrict__ coord,
                 float* __restrict__ magg,
                 float* __restrict__ tsum,
                 const float* __restrict__ deg,
                 const unsigned short* __restrict__ nw1bf,  // [64,32]
                 const float* __restrict__ nb1,
                 const unsigned short* __restrict__ nw2bf,  // [32,32]
                 const float* __restrict__ nb2,
                 int N, int nTiles) {
  __shared__ unsigned short lds_m[WPB][16 * HH];
  const int lane = threadIdx.x & 31;
  const int wid  = threadIdx.x >> 5;
  const int n    = lane & 15;
  const int hi   = (lane & 16) ? 1 : 0;
  const int m    = n;

  Frag w1f[4], w2f[2];
#pragma unroll
  for (int kc = 0; kc < 2; ++kc)
#pragma unroll
    for (int nh = 0; nh < 2; ++nh)
      load_bfrag(&w1f[kc * 2 + nh], nw1bf + kc * 32 * HH, lane, nh);
#pragma unroll
  for (int nh = 0; nh < 2; ++nh) load_bfrag(&w2f[nh], nw2bf, lane, nh);
  const float b1lo = nb1[n], b1hi = nb1[n + 16];
  const float b2lo = nb2[n], b2hi = nb2[n + 16];

  const int stride = gridDim.x * WPB;
  const int iters  = (nTiles + stride - 1) / stride;
  int t = blockIdx.x * WPB + wid;

  for (int it = 0; it < iters; ++it, t += stride) {
    const bool active = (t < nTiles);
    const int v0base = (active ? t : 0) * 16;

    // coord += tsum / max(deg,1); consume-and-zero tsum
    if (lane < 16) {
      int v = v0base + lane;
      if (active && v < N) {
        float idg = frcp(fmaxf(deg[v], 1.0f));
#pragma unroll
        for (int j = 0; j < 3; ++j) {
          coord[v * 3 + j] += tsum[v * 3 + j] * idg;
          tsum[v * 3 + j] = 0.0f;
        }
      }
    }

    int v = v0base + m; if (v >= N) v = N - 1;
    Frag ah, am;
#pragma unroll
    for (int i = 0; i < 16; ++i) {
      int k = ak_of(lane, i);
      ah.u[i] = hbf[v * HH + k];
      am.u[i] = f2bf(magg[v * HH + k]);
    }

    f32x8 c0, c1;
#pragma unroll
    for (int r = 0; r < 8; ++r) { c0[r] = b1lo; c1[r] = b1hi; }
    c0 = wmma_bf16(ah, w1f[0], c0);
    c1 = wmma_bf16(ah, w1f[1], c1);
    c0 = wmma_bf16(am, w1f[2], c0);
    c1 = wmma_bf16(am, w1f[3], c1);

    // consume-and-zero magg (after WMMA has consumed the gathered values)
#pragma unroll
    for (int r = 0; r < 8; ++r) {
      int vv = v0base + r + hi * 8;
      if (active && vv < N) {
        magg[vv * HH + n] = 0.0f;
        magg[vv * HH + 16 + n] = 0.0f;
      }
    }

#pragma unroll
    for (int r = 0; r < 8; ++r) {
      int M = r + hi * 8;
      lds_m[wid][M * HH + n]      = f2bf(silu_f(c0[r]));
      lds_m[wid][M * HH + 16 + n] = f2bf(silu_f(c1[r]));
    }
    __syncthreads();

    Frag a2;
#pragma unroll
    for (int i = 0; i < 16; ++i) a2.u[i] = lds_m[wid][m * HH + ak_of(lane, i)];
    f32x8 d0, d1;
#pragma unroll
    for (int r = 0; r < 8; ++r) { d0[r] = b2lo; d1[r] = b2hi; }
    d0 = wmma_bf16(a2, w2f[0], d0);
    d1 = wmma_bf16(a2, w2f[1], d1);

    // residual: h += hn; refresh bf16 mirror
#pragma unroll
    for (int r = 0; r < 8; ++r) {
      int vv = v0base + r + hi * 8;
      if (active && vv < N) {
        float nv0 = h[vv * HH + n] + d0[r];
        float nv1 = h[vv * HH + 16 + n] + d1[r];
        h[vv * HH + n] = nv0;
        h[vv * HH + 16 + n] = nv1;
        hbf[vv * HH + n] = f2bf(nv0);
        hbf[vv * HH + 16 + n] = f2bf(nv1);
      }
    }
    __syncthreads();
  }
}

// ---------------------------------------------------------------- readout
__global__ void __launch_bounds__(256, 1)
egnn_final_kernel(const unsigned short* __restrict__ hbf,
                  const int* __restrict__ batch,
                  const unsigned short* __restrict__ wobf,  // [32,32] emb_out
                  const float* __restrict__ bo,
                  float* __restrict__ gsum,
                  float* __restrict__ gcnt,
                  int N, int nTiles) {
  const int lane = threadIdx.x & 31;
  const int wid  = threadIdx.x >> 5;
  const int n    = lane & 15;
  const int hi   = (lane & 16) ? 1 : 0;
  const int m    = n;

  Frag wf[2];
#pragma unroll
  for (int nh = 0; nh < 2; ++nh) load_bfrag(&wf[nh], wobf, lane, nh);
  const float blo = bo[n], bhi = bo[n + 16];

  const int stride = gridDim.x * WPB;
  const int iters  = (nTiles + stride - 1) / stride;
  int t = blockIdx.x * WPB + wid;

  for (int it = 0; it < iters; ++it, t += stride) {
    const bool active = (t < nTiles);
    const int v0base = (active ? t : 0) * 16;
    int v = v0base + m; if (v >= N) v = N - 1;
    Frag a;
#pragma unroll
    for (int i = 0; i < 16; ++i) a.u[i] = hbf[v * HH + ak_of(lane, i)];
    f32x8 c0, c1;
#pragma unroll
    for (int r = 0; r < 8; ++r) { c0[r] = blo; c1[r] = bhi; }
    c0 = wmma_bf16(a, wf[0], c0);
    c1 = wmma_bf16(a, wf[1], c1);
#pragma unroll
    for (int r = 0; r < 8; ++r) {
      int vv = v0base + r + hi * 8;
      if (active && vv < N) {
        int g = batch[vv];
        atomicAdd(&gsum[g * ZZ + n],      c0[r]);
        atomicAdd(&gsum[g * ZZ + 16 + n], c1[r]);
      }
    }
    if (lane < 16) {
      int vv = v0base + lane;
      if (active && vv < N) atomicAdd(&gcnt[batch[vv]], 1.0f);
    }
  }
}

__global__ void egnn_head_kernel(const float* __restrict__ gsum,
                                 const float* __restrict__ gcnt,
                                 const float* __restrict__ zmw,
                                 const float* __restrict__ zmb,
                                 const float* __restrict__ zsw,
                                 const float* __restrict__ zsb,
                                 float* __restrict__ out) {
  int idx = blockIdx.x * blockDim.x + threadIdx.x;
  if (idx >= GG * ZZ) return;
  int g = idx >> 5, z = idx & (ZZ - 1);
  float icnt = frcp(fmaxf(gcnt[g], 1.0f));
  float mu = zmb[z], sg = zsb[z];
#pragma unroll 8
  for (int k = 0; k < ZZ; ++k) {
    float hv = gsum[g * ZZ + k] * icnt;
    mu += hv * zmw[k * ZZ + z];
    sg += hv * zsw[k * ZZ + z];
  }
  out[g * ZZ + z] = mu;
  out[GG * ZZ + g * ZZ + z] = (sg > 20.0f) ? sg : log1pf(__expf(sg));
}

// ---------------------------------------------------------------- launch
static inline size_t alignup(size_t x) { return (x + 255) & ~(size_t)255; }
static inline int imin(int a, int b) { return a < b ? a : b; }

extern "C" void kernel_launch(void* const* d_in, const int* in_sizes, int n_in,
                              void* d_out, int out_size, void* d_ws, size_t ws_size,
                              hipStream_t stream) {
  const float* x        = (const float*)d_in[0];
  const float* pos      = (const float*)d_in[1];
  const int*   ei       = (const int*)d_in[2];
  const int*   batch    = (const int*)d_in[3];
  const float* lcell    = (const float*)d_in[5];
  const float* emb_in_w = (const float*)d_in[6];
  const float* emb_in_b = (const float*)d_in[7];
  const float* emb_out_w= (const float*)d_in[8];
  const float* emb_out_b= (const float*)d_in[9];
  const float* edge_w1  = (const float*)d_in[10];
  const float* edge_b1  = (const float*)d_in[11];
  const float* edge_w2  = (const float*)d_in[12];
  const float* edge_b2  = (const float*)d_in[13];
  const float* node_w1  = (const float*)d_in[14];
  const float* node_b1  = (const float*)d_in[15];
  const float* node_w2  = (const float*)d_in[16];
  const float* node_b2  = (const float*)d_in[17];
  const float* coord_w1 = (const float*)d_in[18];
  const float* coord_b1 = (const float*)d_in[19];
  const float* coord_w2 = (const float*)d_in[20];
  const float* coord_b2 = (const float*)d_in[21];
  const float* zmu_w    = (const float*)d_in[22];
  const float* zmu_b    = (const float*)d_in[23];
  const float* zsig_w   = (const float*)d_in[24];
  const float* zsig_b   = (const float*)d_in[25];

  const int N = in_sizes[0];
  const int E = in_sizes[2] / 2;
  const int* row = ei;
  const int* col = ei + E;

  // workspace carve
  char* ws = (char*)d_ws;
  size_t off = 0;
  auto take = [&](size_t bytes) { void* p = ws + off; off = alignup(off + bytes); return p; };
  float* hbuf            = (float*)take((size_t)N * HH * 4);
  unsigned short* hbf    = (unsigned short*)take((size_t)N * HH * 2);
  float* coordbuf        = (float*)take((size_t)N * 3 * 4);
  float* magg            = (float*)take((size_t)N * HH * 4);
  float* tsum            = (float*)take((size_t)N * 3 * 4);
  float* deg             = (float*)take((size_t)N * 4);
  float* gsum            = (float*)take((size_t)GG * ZZ * 4);
  float* gcnt            = (float*)take((size_t)GG * 4);
  unsigned short* ew1bf  = (unsigned short*)take((size_t)NLAYERS * 65 * HH * 2);
  unsigned short* ew2bf  = (unsigned short*)take((size_t)NLAYERS * HH * HH * 2);
  unsigned short* cw1bf  = (unsigned short*)take((size_t)NLAYERS * HH * HH * 2);
  unsigned short* nw1bf  = (unsigned short*)take((size_t)NLAYERS * 2 * HH * HH * 2);
  unsigned short* nw2bf  = (unsigned short*)take((size_t)NLAYERS * HH * HH * 2);
  unsigned short* eobf   = (unsigned short*)take((size_t)HH * ZZ * 2);

  // zero accumulators (capturable)
  hipMemsetAsync(magg, 0, (size_t)N * HH * 4, stream);
  hipMemsetAsync(tsum, 0, (size_t)N * 3 * 4, stream);
  hipMemsetAsync(deg,  0, (size_t)N * 4, stream);
  hipMemsetAsync(gsum, 0, (size_t)GG * ZZ * 4, stream);
  hipMemsetAsync(gcnt, 0, (size_t)GG * 4, stream);

  // bf16 weight conversion
  auto cvt = [&](const float* s, unsigned short* d, int cnt) {
    cvt_bf16_kernel<<<(cnt + 255) / 256, 256, 0, stream>>>(s, d, cnt);
  };
  cvt(edge_w1,  ew1bf, NLAYERS * 65 * HH);
  cvt(edge_w2,  ew2bf, NLAYERS * HH * HH);
  cvt(coord_w1, cw1bf, NLAYERS * HH * HH);
  cvt(node_w1,  nw1bf, NLAYERS * 2 * HH * HH);
  cvt(node_w2,  nw2bf, NLAYERS * HH * HH);
  cvt(emb_out_w, eobf, HH * ZZ);

  // embed + degree
  egnn_embed_kernel<<<((size_t)N * HH + 255) / 256, 256, 0, stream>>>(
      x, pos, emb_in_w, emb_in_b, hbuf, hbf, coordbuf, N);
  egnn_deg_kernel<<<(E + 255) / 256, 256, 0, stream>>>(row, deg, E);

  const int nTilesE = (E + 15) / 16;
  const int nTilesN = (N + 15) / 16;
  const int blocksE = imin((nTilesE + WPB - 1) / WPB, 1728);
  const int blocksN = imin((nTilesN + WPB - 1) / WPB, 512);

  for (int li = 0; li < NLAYERS; ++li) {
    egnn_edge_kernel<<<blocksE, 256, 0, stream>>>(
        hbf, coordbuf, row, col, lcell,
        ew1bf + (size_t)li * 65 * HH,
        edge_w1 + (size_t)li * 65 * HH + 64 * HH,
        edge_b1 + (size_t)li * HH,
        ew2bf + (size_t)li * HH * HH, edge_b2 + (size_t)li * HH,
        cw1bf + (size_t)li * HH * HH, coord_b1 + (size_t)li * HH,
        coord_w2 + (size_t)li * HH, coord_b2 + li,
        magg, tsum, E, nTilesE);
    egnn_node_kernel<<<blocksN, 256, 0, stream>>>(
        hbuf, hbf, coordbuf, magg, tsum, deg,
        nw1bf + (size_t)li * 2 * HH * HH, node_b1 + (size_t)li * HH,
        nw2bf + (size_t)li * HH * HH, node_b2 + (size_t)li * HH,
        N, nTilesN);
  }

  egnn_final_kernel<<<blocksN, 256, 0, stream>>>(
      hbf, batch, eobf, emb_out_b, gsum, gcnt, N, nTilesN);
  egnn_head_kernel<<<(GG * ZZ + 255) / 256, 256, 0, stream>>>(
      gsum, gcnt, zmu_w, zmu_b, zsig_w, zsig_b, (float*)d_out);
}
